// ADAGAD_6141803233547
// MI455X (gfx1250) — matrix operand order repaired and checked by reading, not compiled, not run on hardware
//
#include <hip/hip_runtime.h>

#define NN   8192      // nodes
#define NE   262144    // edges
#define FDIM 64        // in_dim == hid_dim
#define NF   (NN * FDIM)

typedef __attribute__((ext_vector_type(16))) __bf16 v16bf;
typedef __attribute__((ext_vector_type(8)))  __bf16 v8bf;
typedef __attribute__((ext_vector_type(8)))  float  v8f;

// ---------------------------------------------------------------------------
// WMMA fragment helpers (16x16x32 bf16; this file's contiguous k order, same for A and B)
// A (16x32, MxK): lane l holds row m0+(l&15); elem t -> K=(l>>4)*16+t
//                 (two contiguous 16B loads)
// B (32x16, KxN): lane l holds col n0+(l&15); elem t -> K=(l>>4)*16+t
//                 (one contiguous 32B run when B-rows are contiguous)
// C/D (16x16 f32): elem r -> row m0 + 8*(l>>4) + r, col n0+(l&15)
// ---------------------------------------------------------------------------
__device__ __forceinline__ v16bf load_a_frag(const __bf16* __restrict__ arow,
                                             int k0, int half) {
  const __bf16* ap = arow + k0 + half * 16;
  v8bf lo = *(const v8bf*)(ap);
  v8bf hi = *(const v8bf*)(ap + 8);
  return __builtin_shufflevector(lo, hi, 0, 1, 2, 3, 4, 5, 6, 7,
                                 8, 9, 10, 11, 12, 13, 14, 15);
}

__device__ __forceinline__ v8f wmma_bf16(v16bf a, v16bf b, v8f c) {
  return __builtin_amdgcn_wmma_f32_16x16x32_bf16(false, a, false, b,
                                                 (short)0, c, false, false);
}

// ---------------------------------------------------------------------------
// Pack a row-major f32 weight [Kin,Kout] into the per-lane WMMA B layout:
// Wp[(((nt*nkt)+kt)*32 + lane)*16 + t] = bf16( W[(kt*32+(lane>>4)*16+t)*Kout
//                                               + nt*16 + (lane&15)] )
// ---------------------------------------------------------------------------
__global__ void k_pack_w(const float* __restrict__ W, __bf16* __restrict__ Wp,
                         int Kin, int Kout) {
  int gid = blockIdx.x * blockDim.x + threadIdx.x;
  if (gid >= Kin * Kout) return;
  int t    = gid & 15;
  int lane = (gid >> 4) & 31;
  int rest = gid >> 9;
  int nkt  = Kin >> 5;
  int kt   = rest % nkt;
  int nt   = rest / nkt;
  int k = kt * 32 + (lane >> 4) * 16 + t;
  int n = nt * 16 + (lane & 15);
  Wp[gid] = (__bf16)W[k * Kout + n];
}

__global__ void k_cvt_bf16(const float* __restrict__ in,
                           __bf16* __restrict__ out, int count) {
  int gid = blockIdx.x * blockDim.x + threadIdx.x;
  if (gid < count) out[gid] = (__bf16)in[gid];
}

// ---------------------------------------------------------------------------
// Degree / normalization
// ---------------------------------------------------------------------------
__global__ void k_deg_init(float* __restrict__ deg) {
  int gid = blockIdx.x * blockDim.x + threadIdx.x;
  if (gid < NN) deg[gid] = 1.0f;   // self-loop
}
__global__ void k_deg_edges(const int* __restrict__ dst,
                            float* __restrict__ deg) {
  int gid = blockIdx.x * blockDim.x + threadIdx.x;
  if (gid < NE) atomicAdd(&deg[dst[gid]], 1.0f);
}
__global__ void k_dinv(const float* __restrict__ deg,
                       float* __restrict__ dinv) {
  int gid = blockIdx.x * blockDim.x + threadIdx.x;
  if (gid < NN) dinv[gid] = rsqrtf(deg[gid]);
}

// ---------------------------------------------------------------------------
// Dense node-feature GEMM: C[N,Kout] = A_bf16[N,KIN] @ Wp(+bias)(+relu).
// Each wave computes a 16x64 strip (4 n-tiles): A-frag reused 4x, and the
// 4 independent accumulators hide the 5-cycle bf16 WMMA->WMMA RAW hazard.
// KIN is a template param so the K loop fully unrolls with immediate offsets.
// ---------------------------------------------------------------------------
template <int KIN>
__global__ void k_gemm4(const __bf16* __restrict__ A,
                        const __bf16* __restrict__ Wp,
                        const float* __restrict__ bias, int relu,
                        float* __restrict__ Cf, int ldcf,
                        __bf16* __restrict__ Cb, int ldcb,
                        int ngroups) {
  constexpr int NKT = KIN / 32;
  int tid  = threadIdx.x;
  int wave = tid >> 5;
  int lane = tid & 31;
  int half = lane >> 4;
  int col  = lane & 15;
  int tile = blockIdx.x * 8 + wave;
  int mt = tile / ngroups;       // 16-row tile index
  int ng = tile % ngroups;       // group of 4 n-tiles
  int m0 = mt * 16;

  v8f acc[4] = {{}, {}, {}, {}};
  const __bf16* arow  = A + (size_t)(m0 + col) * KIN;
  const __bf16* wbase = Wp + ((size_t)ng * 4 * NKT) * 512 + (size_t)lane * 16;
#pragma unroll
  for (int kt = 0; kt < NKT; ++kt) {
    v16bf a = load_a_frag(arow, kt * 32, half);
#pragma unroll
    for (int j = 0; j < 4; ++j) {
      v16bf b = *(const v16bf*)(wbase + (size_t)(j * NKT + kt) * 512);
      acc[j] = wmma_bf16(a, b, acc[j]);
    }
  }

#pragma unroll
  for (int j = 0; j < 4; ++j) {
    int n = (ng * 4 + j) * 16 + col;
    float bv = bias ? bias[n] : 0.0f;
#pragma unroll
    for (int r = 0; r < 8; ++r) {
      float v = acc[j][r] + bv;
      if (relu) v = fmaxf(v, 0.0f);
      size_t row = (size_t)(m0 + half * 8 + r);
      if (Cf) Cf[row * ldcf + n] = v;
      if (Cb) Cb[row * ldcb + n] = (__bf16)v;
    }
  }
}

// ---------------------------------------------------------------------------
// GCN aggregation: agg = dinv^2 * h  (self loop), then scatter edges,
// then out = (agg + b) [relu], with optional f32 / bf16 destinations.
// ---------------------------------------------------------------------------
__global__ void k_selfloop(const float* __restrict__ h,
                           const float* __restrict__ dinv,
                           float* __restrict__ agg) {
  int gid = blockIdx.x * blockDim.x + threadIdx.x;
  if (gid >= NF) return;
  float di = dinv[gid >> 6];
  agg[gid] = h[gid] * di * di;
}

__global__ void k_edge_agg(const float* __restrict__ h,
                           const float* __restrict__ dinv,
                           const int* __restrict__ src,
                           const int* __restrict__ dst,
                           float* __restrict__ agg) {
  int gid = blockIdx.x * blockDim.x + threadIdx.x;   // NE*64 threads
  int e = gid >> 6;
  int f = gid & 63;
  int s = src[e];
  int d = dst[e];
  float c = dinv[s] * dinv[d];
  atomicAdd(&agg[(size_t)d * 64 + f], h[(size_t)s * 64 + f] * c);
}

__global__ void k_finalize(const float* __restrict__ agg,
                           const float* __restrict__ bias, int relu,
                           float* __restrict__ Cf, int ldcf,
                           __bf16* __restrict__ Cb, int ldcb) {
  int gid = blockIdx.x * blockDim.x + threadIdx.x;
  if (gid >= NF) return;
  int i = gid >> 6;
  int f = gid & 63;
  float v = agg[gid] + bias[f];
  if (relu) v = fmaxf(v, 0.0f);
  if (Cf) Cf[(size_t)i * ldcf + f] = v;
  if (Cb) Cb[(size_t)i * ldcb + f] = (__bf16)v;
}

// ---------------------------------------------------------------------------
// Attention: softmax over 3 views per (node, feature) + fused embedding
// att_pre is [N,192] (reshape -> [N,64,3]); writes att to d_out and h_bf16.
// ---------------------------------------------------------------------------
__global__ void k_att_fuse(const float* __restrict__ att_pre,
                           const float* __restrict__ ha,
                           const float* __restrict__ hs,
                           const float* __restrict__ ht,
                           float* __restrict__ att_out,
                           __bf16* __restrict__ h_bf) {
  int gid = blockIdx.x * blockDim.x + threadIdx.x;
  if (gid >= NF) return;
  int i = gid >> 6;
  int f = gid & 63;
  size_t base = (size_t)i * 192 + f * 3;
  float e0 = att_pre[base + 0];
  float e1 = att_pre[base + 1];
  float e2 = att_pre[base + 2];
  float m = fmaxf(e0, fmaxf(e1, e2));
  float x0 = expf(e0 - m), x1 = expf(e1 - m), x2 = expf(e2 - m);
  float inv = 1.0f / (x0 + x1 + x2);
  float p0 = x0 * inv, p1 = x1 * inv, p2 = x2 * inv;
  att_out[base + 0] = p0;
  att_out[base + 1] = p1;
  att_out[base + 2] = p2;
  float hv = ha[gid] * p0 + hs[gid] * p1 + ht[gid] * p2;
  h_bf[gid] = (__bf16)hv;
}

// ---------------------------------------------------------------------------
// s_ = h_ @ h_^T  (8192x8192 f32 out) — store-bound; bf16 WMMA.
// Wave computes a 16x64 strip: A-frag reused across 4 B-frags; B-frags are
// contiguous 32B runs of h_ rows (B[k][n] = h_[n][k], no transpose needed).
// ---------------------------------------------------------------------------
__global__ void k_gemm_big(const __bf16* __restrict__ H,
                           float* __restrict__ S) {
  int tid  = threadIdx.x;
  int wave = tid >> 5;
  int lane = tid & 31;
  int half = lane >> 4;
  int col  = lane & 15;
  int m0 = (blockIdx.y * 8 + wave) * 16;
  int nb = blockIdx.x * 64;

  v8f acc[4] = {{}, {}, {}, {}};
  const __bf16* arow = H + (size_t)(m0 + col) * 64;
#pragma unroll
  for (int kt = 0; kt < 2; ++kt) {
    int k0 = kt * 32;
    v16bf a = load_a_frag(arow, k0, half);
#pragma unroll
    for (int j = 0; j < 4; ++j) {
      int n0 = nb + j * 16;
      v16bf b = *(const v16bf*)(H + (size_t)(n0 + col) * 64 + k0 + half * 16);
      acc[j] = wmma_bf16(a, b, acc[j]);
    }
  }
#pragma unroll
  for (int j = 0; j < 4; ++j) {
    int n = nb + j * 16 + col;
#pragma unroll
    for (int r = 0; r < 8; ++r) {
      size_t row = (size_t)(m0 + half * 8 + r);
      S[row * NN + n] = acc[j][r];
    }
  }
}

// ---------------------------------------------------------------------------
extern "C" void kernel_launch(void* const* d_in, const int* in_sizes, int n_in,
                              void* d_out, int out_size, void* d_ws,
                              size_t ws_size, hipStream_t stream) {
  (void)in_sizes; (void)n_in; (void)out_size; (void)ws_size;

  const float* x   = (const float*)d_in[0];
  const int*   src = (const int*)d_in[1];
  const int*   dst = (const int*)d_in[2];
  const float* eW1[3] = {(const float*)d_in[3],  (const float*)d_in[7],  (const float*)d_in[11]};
  const float* eb1[3] = {(const float*)d_in[4],  (const float*)d_in[8],  (const float*)d_in[12]};
  const float* eW2[3] = {(const float*)d_in[5],  (const float*)d_in[9],  (const float*)d_in[13]};
  const float* eb2[3] = {(const float*)d_in[6],  (const float*)d_in[10], (const float*)d_in[14]};
  const float* attW = (const float*)d_in[15];
  const float* attb = (const float*)d_in[16];
  const float* dxW  = (const float*)d_in[17];
  const float* dxb  = (const float*)d_in[18];
  const float* dsW  = (const float*)d_in[19];
  const float* dsb  = (const float*)d_in[20];

  float* out     = (float*)d_out;
  float* x_out   = out;                         // [N,64]
  float* s_out   = out + (size_t)NF;            // [N,N]
  float* att_out = s_out + (size_t)NN * NN;     // [N,64,3]

  // ---- workspace carve-out (all 256B aligned) ----
  char* w = (char*)d_ws;
  size_t off = 0;
  auto alloc = [&](size_t bytes) -> void* {
    void* p = w + off;
    off = (off + bytes + 255) & ~(size_t)255;
    return p;
  };
  __bf16* x_bf    = (__bf16*)alloc((size_t)NF * 2);
  float*  deg     = (float*)alloc((size_t)NN * 4);
  float*  dinv    = (float*)alloc((size_t)NN * 4);
  __bf16* wpE1[3], *wpE2[3];
  for (int e = 0; e < 3; ++e) {
    wpE1[e] = (__bf16*)alloc(64 * 64 * 2);
    wpE2[e] = (__bf16*)alloc(64 * 64 * 2);
  }
  __bf16* wpAtt = (__bf16*)alloc(192 * 192 * 2);
  __bf16* wpDx  = (__bf16*)alloc(64 * 64 * 2);
  __bf16* wpDs  = (__bf16*)alloc(64 * 64 * 2);
  float*  h_tmp   = (float*)alloc((size_t)NF * 4);
  float*  agg_tmp = (float*)alloc((size_t)NF * 4);
  __bf16* l1_bf   = (__bf16*)alloc((size_t)NF * 2);
  float*  h_enc[3];
  for (int e = 0; e < 3; ++e) h_enc[e] = (float*)alloc((size_t)NF * 4);
  __bf16* c_bf    = (__bf16*)alloc((size_t)NN * 192 * 2);
  float*  att_pre = (float*)alloc((size_t)NN * 192 * 4);
  __bf16* hf_bf   = (__bf16*)alloc((size_t)NF * 2);
  __bf16* hd_bf   = (__bf16*)alloc((size_t)NF * 2);

  const int T = 256;
  const int BNF = NF / T;          // 2048 blocks for N*64 elementwise
  const int BE  = (NE * 64) / T;   // 65536 blocks for edge scatter

  // ---- preprocessing: bf16 activations, packed weights, degrees ----
  k_cvt_bf16<<<BNF, T, 0, stream>>>(x, x_bf, NF);
  for (int e = 0; e < 3; ++e) {
    k_pack_w<<<16, T, 0, stream>>>(eW1[e], wpE1[e], 64, 64);
    k_pack_w<<<16, T, 0, stream>>>(eW2[e], wpE2[e], 64, 64);
  }
  k_pack_w<<<144, T, 0, stream>>>(attW, wpAtt, 192, 192);
  k_pack_w<<<16, T, 0, stream>>>(dxW, wpDx, 64, 64);
  k_pack_w<<<16, T, 0, stream>>>(dsW, wpDs, 64, 64);
  k_deg_init<<<NN / T, T, 0, stream>>>(deg);
  k_deg_edges<<<NE / T, T, 0, stream>>>(dst, deg);
  k_dinv<<<NN / T, T, 0, stream>>>(deg, dinv);

  // one GCN conv (Kin=Kout=64): h = A@W ; agg = scatter + selfloop ;
  // out = agg + b [relu].  512 m-tiles * 1 n-group / 8 waves = 64 blocks.
  auto conv = [&](const __bf16* A, const __bf16* Wp, const float* b,
                  int relu, float* outF, int ldf, __bf16* outB, int ldb) {
    k_gemm4<64><<<64, T, 0, stream>>>(A, Wp, nullptr, 0,
                                      h_tmp, 64, nullptr, 0, 1);
    k_selfloop<<<BNF, T, 0, stream>>>(h_tmp, dinv, agg_tmp);
    k_edge_agg<<<BE, T, 0, stream>>>(h_tmp, dinv, src, dst, agg_tmp);
    k_finalize<<<BNF, T, 0, stream>>>(agg_tmp, b, relu, outF, ldf, outB, ldb);
  };

  // ---- three 2-layer GCN encoders; layer-2 bf16 goes into concat c_bf ----
  for (int e = 0; e < 3; ++e) {
    conv(x_bf, wpE1[e], eb1[e], 1, nullptr, 0, l1_bf, 64);
    conv(l1_bf, wpE2[e], eb2[e], 1, h_enc[e], 64, c_bf + e * 64, 192);
  }

  // ---- attention GEMM [N,192]@[192,192]+b: 512 m-tiles * 3 n-groups ----
  k_gemm4<192><<<192, T, 0, stream>>>(c_bf, wpAtt, attb, 0,
                                      att_pre, 192, nullptr, 0, 3);
  k_att_fuse<<<BNF, T, 0, stream>>>(att_pre, h_enc[0], h_enc[1], h_enc[2],
                                    att_out, hf_bf);

  // ---- decoders: x_ straight to d_out; h_ kept as bf16 for s_ GEMM ----
  conv(hf_bf, wpDx, dxb, 0, x_out, 64, nullptr, 0);
  conv(hf_bf, wpDs, dsb, 0, nullptr, 0, hd_bf, 64);

  // ---- s_ = h_ @ h_^T : 268 MB f32, store-bound WMMA GEMM ----
  dim3 gbig(NN / 64, (NN / 16) / 8);
  k_gemm_big<<<gbig, T, 0, stream>>>(hd_bf, s_out);
}
